// PentachoronCantorCollective_46523085750376
// MI455X (gfx1250) — compile-verified, matrix operand order checked by reading, not run on hardware
//
#include <hip/hip_runtime.h>

#define VOCAB 32000
#define DIM 512
#define NUM_HEADS 8
#define HEAD_DIM 64
#define BATCH 2
#define SEQ 2048
#define KNBR 64
#define NTOK (BATCH * SEQ)     // 4096
#define QKV_N (3 * DIM)        // 1536
#define EPSV 1e-06f

typedef __attribute__((ext_vector_type(16))) __bf16 v16bf;
typedef __attribute__((ext_vector_type(8)))  float  v8f;

__device__ __forceinline__ __bf16 to_bf16(float f) {
    union { float f; unsigned u; } c; c.f = f;
    unsigned r = c.u + 0x7FFFu + ((c.u >> 16) & 1u);   // round-to-nearest-even
    unsigned short h = (unsigned short)(r >> 16);
    return __builtin_bit_cast(__bf16, h);
}
__device__ __forceinline__ float from_bf16(__bf16 b) {
    unsigned short h = __builtin_bit_cast(unsigned short, b);
    union { unsigned u; float f; } c; c.u = ((unsigned)h) << 16;
    return c.f;
}
__device__ __forceinline__ float wave_sum(float v) {
    #pragma unroll
    for (int off = 16; off > 0; off >>= 1) v += __shfl_xor(v, off, 32);
    return v;
}
__device__ __forceinline__ float wave_max(float v) {
    #pragma unroll
    for (int off = 16; off > 0; off >>= 1) v = fmaxf(v, __shfl_xor(v, off, 32));
    return v;
}
__device__ __forceinline__ float sigmoidf(float x) { return 1.0f / (1.0f + __expf(-x)); }

// ---------------------------------------------------------------------------
// Kernel 1: Cantor fingerprint per vocab entry. One wave32 per token.
// Each lane owns 16 contiguous dims of each of the 5 vertices.
// ---------------------------------------------------------------------------
__global__ void cantor_kernel(const float* __restrict__ penta, float* __restrict__ coords) {
    int wid  = (blockIdx.x * blockDim.x + threadIdx.x) >> 5;
    int lane = threadIdx.x & 31;
    if (wid >= VOCAB) return;

    const float* base = penta + (size_t)wid * 5 * DIM + lane * 16;
    float p[5][16];
    #pragma unroll
    for (int i = 0; i < 5; ++i) {
        #pragma unroll
        for (int d = 0; d < 16; ++d) p[i][d] = base[i * DIM + d];
    }

    float sums[15];
    int idx = 0;
    #pragma unroll
    for (int i = 0; i < 5; ++i)
        #pragma unroll
        for (int j = i + 1; j < 5; ++j) {
            float a = 0.f;
            #pragma unroll
            for (int d = 0; d < 16; ++d) { float t = p[i][d] - p[j][d]; a += t * t; }
            sums[idx++] = a;
        }
    #pragma unroll
    for (int i = 0; i < 5; ++i) {
        float a = 0.f;
        #pragma unroll
        for (int d = 0; d < 16; ++d) {
            float m = (p[0][d] + p[1][d] + p[2][d] + p[3][d] + p[4][d]) * 0.2f;
            float t = p[i][d] - m; a += t * t;
        }
        sums[idx++] = a;
    }
    #pragma unroll
    for (int t = 0; t < 15; ++t) sums[t] = wave_sum(sums[t]);

    // scalar tail (all lanes compute; lane 0 stores)
    float dq[5][5];
    idx = 0;
    #pragma unroll
    for (int i = 0; i < 5; ++i) dq[i][i] = 0.f;
    #pragma unroll
    for (int i = 0; i < 5; ++i)
        #pragma unroll
        for (int j = i + 1; j < 5; ++j) { dq[i][j] = sums[idx]; dq[j][i] = sums[idx]; ++idx; }

    float edge[10], mean_edge = 0.f;
    #pragma unroll
    for (int t = 0; t < 10; ++t) { edge[t] = sqrtf(sums[t]); mean_edge += edge[t]; }
    mean_edge *= 0.1f;
    float var_e = 0.f;
    #pragma unroll
    for (int t = 0; t < 10; ++t) { float d = edge[t] - mean_edge; var_e += d * d; }
    float std_edge = sqrtf(var_e / 9.0f);

    float cd[5], mc = 0.f;
    #pragma unroll
    for (int i = 0; i < 5; ++i) { cd[i] = sqrtf(sums[10 + i]); mc += cd[i]; }
    mc *= 0.2f;
    float var_c = 0.f;
    #pragma unroll
    for (int i = 0; i < 5; ++i) { float d = cd[i] - mc; var_c += d * d; }
    float spread = sqrtf(var_c / 4.0f);

    // Cayley-Menger 6x6 determinant (Gaussian elimination w/ partial pivot)
    float A[6][6];
    A[0][0] = 0.f;
    #pragma unroll
    for (int j = 1; j < 6; ++j) { A[0][j] = 1.f; A[j][0] = 1.f; }
    #pragma unroll
    for (int i = 0; i < 5; ++i)
        #pragma unroll
        for (int j = 0; j < 5; ++j) A[1 + i][1 + j] = dq[i][j];

    float det = 1.f;
    for (int c = 0; c < 6; ++c) {
        int piv = c; float best = fabsf(A[c][c]);
        for (int r = c + 1; r < 6; ++r) {
            float a = fabsf(A[r][c]);
            if (a > best) { best = a; piv = r; }
        }
        if (piv != c) {
            for (int k = 0; k < 6; ++k) { float t = A[c][k]; A[c][k] = A[piv][k]; A[piv][k] = t; }
            det = -det;
        }
        float pv = A[c][c];
        det *= pv;
        if (pv != 0.f) {
            float inv = 1.0f / pv;
            for (int r = c + 1; r < 6; ++r) {
                float f = A[r][c] * inv;
                for (int k = c; k < 6; ++k) A[r][k] -= f * A[c][k];
            }
        }
    }
    float volume = sqrtf(fmaxf(-det / 9216.0f, 0.0f));
    float vn = sigmoidf(volume * 10.0f);
    float er = sigmoidf(std_edge / (mean_edge + 1e-06f));
    float sn = sigmoidf(spread);
    float x = fminf(fmaxf(vn * 0.4f + er * 0.3f + sn * 0.3f, EPSV), 1.0f - EPSV);
    float bump = (vn + er + sn) * 0.01f;

    float cv = 0.f, factor = 0.5f;
    #pragma unroll
    for (int it = 0; it < 8; ++it) {
        float xs = x * 3.0f;
        float digit = floorf(xs);
        float frac = xs - digit;
        cv += (digit == 2.0f) ? factor : 0.0f;
        x = fminf(fmaxf(frac + bump, EPSV), 1.0f - EPSV);
        factor *= 0.5f;
    }
    if (lane == 0) coords[wid] = fminf(fmaxf(cv, 0.0f), 1.0f);
}

// ---------------------------------------------------------------------------
// Kernel 2: per-row top-64 nearest neighbors by |c_i - c_j|. One block per row.
// Tie-break: smaller index wins (matches jax top_k on -dist).
// ---------------------------------------------------------------------------
__global__ void topk_kernel(const float* __restrict__ coords, const int* __restrict__ token_ids,
                            int* __restrict__ routes) {
    __shared__ float dist[SEQ];
    __shared__ float redv[256];
    __shared__ int   redi[256];
    int s = blockIdx.x;
    float cs = coords[token_ids[s]];
    for (int j = threadIdx.x; j < SEQ; j += 256)
        dist[j] = fabsf(cs - coords[token_ids[j]]);
    __syncthreads();

    for (int k = 0; k < KNBR; ++k) {
        float best = __builtin_inff(); int bi = 0x7fffffff;
        for (int j = threadIdx.x; j < SEQ; j += 256) {
            float d = dist[j];
            if (d < best || (d == best && j < bi)) { best = d; bi = j; }
        }
        redv[threadIdx.x] = best; redi[threadIdx.x] = bi;
        __syncthreads();
        for (int off = 128; off > 0; off >>= 1) {
            if ((int)threadIdx.x < off) {
                float ov = redv[threadIdx.x + off]; int oi = redi[threadIdx.x + off];
                if (ov < redv[threadIdx.x] || (ov == redv[threadIdx.x] && oi < redi[threadIdx.x])) {
                    redv[threadIdx.x] = ov; redi[threadIdx.x] = oi;
                }
            }
            __syncthreads();
        }
        if (threadIdx.x == 0) { routes[s * KNBR + k] = redi[0]; dist[redi[0]] = __builtin_inff(); }
        __syncthreads();
    }
}

// ---------------------------------------------------------------------------
// Conversion kernels
// ---------------------------------------------------------------------------
__global__ void f32_to_bf16_kernel(const float* __restrict__ in, __bf16* __restrict__ out, int n) {
    int i = blockIdx.x * blockDim.x + threadIdx.x;
    if (i < n) out[i] = to_bf16(in[i]);
}
// in[rows][cols] (row-major) -> out[cols][rows] bf16 (store B^T so WMMA B-frag reads are contiguous)
__global__ void transpose_to_bf16_kernel(const float* __restrict__ in, __bf16* __restrict__ out,
                                         int rows, int cols) {
    int i = blockIdx.x * blockDim.x + threadIdx.x;
    if (i < rows * cols) {
        int r = i / cols, c = i % cols;
        out[(size_t)c * rows + r] = to_bf16(in[i]);
    }
}

// ---------------------------------------------------------------------------
// Kernel 4/6: bf16 WMMA GEMM, register-blocked 16x64 per wave.
// C[M,N] = A[M,K] * B[K,N] + bias, B given as BT[N,K].
// Each wave owns one 16-row strip and FOUR adjacent 16-col tiles: the A
// fragment is loaded once per k-step and reused across 4 WMMAs (4 accumulators)
// -> 4x fewer A-fragment loads, WMMA:VMEM ratio 4:5 per k-step.
// A-frag: lanes 0-15 row M=r K[k0..k0+15]; lanes 16-31 same row, K[k0+16..+31].
// B-frag: mirrored with N on lanes (reads BT rows contiguously).
// ---------------------------------------------------------------------------
__global__ void gemm_bf16_wmma_kernel(const __bf16* __restrict__ Abuf, const __bf16* __restrict__ BT,
                                      const float* __restrict__ bias,
                                      __bf16* __restrict__ out_bf, float* __restrict__ out_f32,
                                      int M, int N, int K) {
    int wave = (blockIdx.x * blockDim.x + threadIdx.x) >> 5;
    int lane = threadIdx.x & 31;
    int ncg = N >> 6;                 // 64-wide column groups
    int tm = (wave / ncg) << 4;
    int tn = (wave % ncg) << 6;
    if (tm >= M) return;              // uniform per-wave; never taken with exact grids

    int hlf = lane >> 4;              // 0: K low half, 1: K high half
    int r   = lane & 15;              // row within tile (A) / col within tile (B)
    const __bf16* pa  = Abuf + (size_t)(tm + r) * K + hlf * 16;
    const __bf16* pb0 = BT   + (size_t)(tn + r) * K + hlf * 16;
    size_t bstride16 = (size_t)16 * K;   // 16 BT rows

    v8f acc0 = {}, acc1 = {}, acc2 = {}, acc3 = {};
    #pragma unroll 2
    for (int k0 = 0; k0 < K; k0 += 32) {
        __builtin_prefetch(pa + k0 + 256, 0, 1);
        __builtin_prefetch(pb0 + k0 + 256, 0, 1);
        v16bf a  = *(const v16bf*)(pa + k0);
        v16bf b0 = *(const v16bf*)(pb0 + k0);
        v16bf b1 = *(const v16bf*)(pb0 + bstride16 + k0);
        v16bf b2 = *(const v16bf*)(pb0 + 2 * bstride16 + k0);
        v16bf b3 = *(const v16bf*)(pb0 + 3 * bstride16 + k0);
        acc0 = __builtin_amdgcn_wmma_f32_16x16x32_bf16(false, a, false, b0, (short)0, acc0, false, false);
        acc1 = __builtin_amdgcn_wmma_f32_16x16x32_bf16(false, a, false, b1, (short)0, acc1, false, false);
        acc2 = __builtin_amdgcn_wmma_f32_16x16x32_bf16(false, a, false, b2, (short)0, acc2, false, false);
        acc3 = __builtin_amdgcn_wmma_f32_16x16x32_bf16(false, a, false, b3, (short)0, acc3, false, false);
    }

    // D layout: VGPR i -> row tm + i + 8*hlf, col tile_n + (lane&15)
    int m0 = tm + (hlf ? 8 : 0);
    v8f accs[4] = {acc0, acc1, acc2, acc3};
    #pragma unroll
    for (int j = 0; j < 4; ++j) {
        int col = tn + j * 16 + r;
        float bv = bias[col];
        #pragma unroll
        for (int i = 0; i < 8; ++i) {
            float v = accs[j][i] + bv;
            size_t o = (size_t)(m0 + i) * N + col;
            if (out_f32) out_f32[o] = v;
            else         out_bf[o]  = to_bf16(v);
        }
    }
}

// ---------------------------------------------------------------------------
// Kernel 5: gathered-neighbor attention. One wave32 per (b,h,q).
// qkv layout: [token][3*DIM] with token = b*SEQ+s, part offsets {0,512,1024}, head h at h*64.
// ---------------------------------------------------------------------------
__global__ void attn_kernel(const __bf16* __restrict__ qkv, const int* __restrict__ routes,
                            __bf16* __restrict__ out) {
    int wid    = (blockIdx.x * blockDim.x + threadIdx.x) >> 5;
    int lane   = threadIdx.x & 31;
    int wlocal = threadIdx.x >> 5;

    int s = wid & (SEQ - 1);
    int h = (wid >> 11) & (NUM_HEADS - 1);
    int b = wid >> 14;

    __shared__ float qs[4][HEAD_DIM];
    __shared__ float aw[4][KNBR];
    __shared__ int   rt[4][KNBR];

    size_t qbase = (size_t)(b * SEQ + s) * QKV_N + h * HEAD_DIM;
    qs[wlocal][lane * 2]     = from_bf16(qkv[qbase + lane * 2]);
    qs[wlocal][lane * 2 + 1] = from_bf16(qkv[qbase + lane * 2 + 1]);
    rt[wlocal][lane * 2]     = routes[s * KNBR + lane * 2];
    rt[wlocal][lane * 2 + 1] = routes[s * KNBR + lane * 2 + 1];
    __syncthreads();

    // two neighbors per lane: scores
    float sc[2];
    #pragma unroll
    for (int t = 0; t < 2; ++t) {
        int n = lane * 2 + t;
        int j = rt[wlocal][n];
        const __bf16* kp = qkv + (size_t)(b * SEQ + j) * QKV_N + DIM + h * HEAD_DIM;
        float a = 0.f;
        #pragma unroll
        for (int d = 0; d < HEAD_DIM; ++d) a += qs[wlocal][d] * from_bf16(kp[d]);
        sc[t] = a * 0.125f;  // 1/sqrt(64)
    }
    float m = wave_max(fmaxf(sc[0], sc[1]));
    float e0 = __expf(sc[0] - m), e1 = __expf(sc[1] - m);
    float ssum = wave_sum(e0 + e1);
    float inv = 1.0f / ssum;
    aw[wlocal][lane * 2]     = e0 * inv;
    aw[wlocal][lane * 2 + 1] = e1 * inv;
    __syncthreads();

    // two output dims per lane
    int d0 = lane * 2;
    float o0 = 0.f, o1 = 0.f;
    #pragma unroll 8
    for (int n = 0; n < KNBR; ++n) {
        int j = rt[wlocal][n];
        float a = aw[wlocal][n];
        const __bf16* vp = qkv + (size_t)(b * SEQ + j) * QKV_N + 2 * DIM + h * HEAD_DIM;
        o0 += a * from_bf16(vp[d0]);
        o1 += a * from_bf16(vp[d0 + 1]);
    }
    size_t obase = (size_t)(b * SEQ + s) * DIM + h * HEAD_DIM;
    out[obase + d0]     = to_bf16(o0);
    out[obase + d0 + 1] = to_bf16(o1);
}

// ---------------------------------------------------------------------------
// Host launcher
// ---------------------------------------------------------------------------
extern "C" void kernel_launch(void* const* d_in, const int* in_sizes, int n_in,
                              void* d_out, int out_size, void* d_ws, size_t ws_size,
                              hipStream_t stream) {
    const float* pentachora = (const float*)d_in[0];
    const float* x          = (const float*)d_in[1];
    const int*   token_ids  = (const int*)d_in[2];
    const float* Wqkv       = (const float*)d_in[3];
    const float* bqkv       = (const float*)d_in[4];
    const float* Wout       = (const float*)d_in[5];
    const float* bout       = (const float*)d_in[6];
    float* out = (float*)d_out;

    // workspace carve-up (256B aligned)
    char* ws = (char*)d_ws;
    size_t off = 0;
    auto carve = [&](size_t bytes) {
        char* p = ws + off;
        off = (off + bytes + 255) & ~(size_t)255;
        return p;
    };
    float*  coords   = (float*)carve((size_t)VOCAB * 4);
    int*    routes   = (int*)carve((size_t)SEQ * KNBR * 4);
    __bf16* x_bf     = (__bf16*)carve((size_t)NTOK * DIM * 2);
    __bf16* wqkvT    = (__bf16*)carve((size_t)QKV_N * DIM * 2);
    __bf16* woutT    = (__bf16*)carve((size_t)DIM * DIM * 2);
    __bf16* qkv_bf   = (__bf16*)carve((size_t)NTOK * QKV_N * 2);
    __bf16* attn_bf  = (__bf16*)carve((size_t)NTOK * DIM * 2);
    (void)ws_size; (void)n_in; (void)in_sizes; (void)out_size;

    // 1) fingerprints: 32000 waves, 4 waves/block
    cantor_kernel<<<VOCAB / 4, 128, 0, stream>>>(pentachora, coords);

    // 2) top-64 routes per query row
    topk_kernel<<<SEQ, 256, 0, stream>>>(coords, token_ids, routes);

    // 3) bf16 conversions (+ weight transposes for WMMA B-frag layout)
    {
        int n = NTOK * DIM;
        f32_to_bf16_kernel<<<(n + 255) / 256, 256, 0, stream>>>(x, x_bf, n);
    }
    {
        int n = DIM * QKV_N;
        transpose_to_bf16_kernel<<<(n + 255) / 256, 256, 0, stream>>>(Wqkv, wqkvT, DIM, QKV_N);
    }
    {
        int n = DIM * DIM;
        transpose_to_bf16_kernel<<<(n + 255) / 256, 256, 0, stream>>>(Wout, woutT, DIM, DIM);
    }

    // 4) QKV projection: [4096,512] x [512,1536] -> bf16 qkv
    //    waves = (M/16) * (N/64) = 256 * 24 = 6144; 4 waves/block -> 1536 blocks
    {
        int waves = (NTOK / 16) * (QKV_N / 64);
        gemm_bf16_wmma_kernel<<<waves / 4, 128, 0, stream>>>(
            x_bf, wqkvT, bqkv, qkv_bf, nullptr, NTOK, QKV_N, DIM);
    }

    // 5) gathered attention: 32768 waves, 4/block
    attn_kernel<<<(BATCH * NUM_HEADS * SEQ) / 4, 128, 0, stream>>>(qkv_bf, routes, attn_bf);

    // 6) output projection: [4096,512] x [512,512] + bout -> f32 d_out
    //    waves = 256 * 8 = 2048 -> 512 blocks
    {
        int waves = (NTOK / 16) * (DIM / 64);
        gemm_bf16_wmma_kernel<<<waves / 4, 128, 0, stream>>>(
            attn_bf, woutT, bout, nullptr, out, NTOK, DIM, DIM);
    }
}
